// FrequencyTimeAttention_63385127354620
// MI455X (gfx1250) — matrix operand order repaired and checked by reading, not compiled, not run on hardware
//
#include <hip/hip_runtime.h>
#include <hip/hip_bf16.h>
#include <stdint.h>

#define DIMX 512
#define HEADS 8
#define HEAD_DIM 64
#define BATCH 8
#define SEQ 512
#define PTAB 64

typedef __attribute__((ext_vector_type(16))) __bf16 v16bf;
typedef __attribute__((ext_vector_type(8)))  float  v8f;
typedef unsigned int __attribute__((ext_vector_type(4))) u32x4;
typedef int  __attribute__((ext_vector_type(4))) i32x4;
typedef int  __attribute__((ext_vector_type(8))) i32x8;

// ---------------- WMMA fragment loaders (contiguous k = 16h + e order, wave32) -------
// A-fragment 16x32 (MxK): lane half h, row m=lane&15; element e -> K = 16*h + e
__device__ __forceinline__ v16bf load_a_bf16(const __bf16* src, int ld, int row0, int k0, int lane) {
    int half = lane >> 4, m = lane & 15;
    const __bf16* p = src + (long)(row0 + m) * ld + k0;
    v16bf a;
#pragma unroll
    for (int e = 0; e < 16; ++e) a[e] = p[e + 16 * half];
    return a;
}
// A-fragment from a wave-private LDS score tile (fp32, row0 = 0)
__device__ __forceinline__ v16bf load_a_lds_f32(const float* src, int ld, int k0, int lane) {
    int half = lane >> 4, m = lane & 15;
    const float* p = src + (long)m * ld + k0;
    v16bf a;
#pragma unroll
    for (int e = 0; e < 16; ++e) a[e] = (__bf16)p[e + 16 * half];
    return a;
}
// B-fragment where B[k][n] = src[n][k] (transposed source): contiguous along k
__device__ __forceinline__ v16bf load_bT_bf16(const __bf16* src, int ld, int k0, int n0, int lane) {
    int half = lane >> 4, n = lane & 15;
    const __bf16* p = src + (long)(n0 + n) * ld + k0 + 16 * half;
    v16bf b;
#pragma unroll
    for (int e = 0; e < 16; ++e) b[e] = p[e];
    return b;
}
__device__ __forceinline__ v16bf load_bT_f32(const float* src, int ld, int k0, int n0, int lane) {
    int half = lane >> 4, n = lane & 15;
    const float* p = src + (long)(n0 + n) * ld + k0 + 16 * half;
    v16bf b;
#pragma unroll
    for (int e = 0; e < 16; ++e) b[e] = (__bf16)p[e];
    return b;
}
// B-fragment from LDS-staged row-major bf16 tile [K rows][ld cols]
__device__ __forceinline__ v16bf load_b_lds_bf16(const __bf16* src, int ld, int n0, int lane) {
    int half = lane >> 4, n = lane & 15;
    const __bf16* p = src + (long)(16 * half) * ld + (n0 + n);
    v16bf b;
#pragma unroll
    for (int e = 0; e < 16; ++e) b[e] = p[(long)e * ld];
    return b;
}

#define WMMA_BF16(c, a, b) \
    __builtin_amdgcn_wmma_f32_16x16x32_bf16(false, (a), false, (b), (short)0, (c), false, false)

// ---- TDM: 2-D bf16 tile [tile1 rows x tile0 cols] global -> LDS -----------
// D# per CDNA5 ISA 8.3/8.4: group0 {count=1,type=2,lds,global}, group1 {dims}
__device__ __forceinline__ void tdm_load_2d_bf16(uint32_t lds_addr, const void* gptr,
                                                 uint32_t tensor0, uint32_t tensor1,
                                                 uint32_t stride0,
                                                 uint32_t tile0, uint32_t tile1) {
    uint64_t ga = (uint64_t)(uintptr_t)gptr;
    u32x4 g0;
    g0[0] = 1u;                                       // count=1, user mode
    g0[1] = lds_addr;                                 // lds_addr [63:32]
    g0[2] = (uint32_t)(ga & 0xffffffffu);             // global_addr [95:64]
    g0[3] = (uint32_t)((ga >> 32) & 0x1ffffffu) | (2u << 30); // addr[56:32], type=2
    i32x8 g1;
    g1[0] = (int)(1u << 16);                          // data_size=1 (2 bytes), mask=0
    g1[1] = (int)((tensor0 & 0xffffu) << 16);         // tensor_dim0[15:0] @ [63:48]
    g1[2] = (int)((tensor0 >> 16) | ((tensor1 & 0xffffu) << 16)); // td0 hi | td1 lo
    g1[3] = (int)((tensor1 >> 16) | (tile0 << 16));   // td1 hi | tile_dim0
    g1[4] = (int)(tile1);                             // tile_dim1 | tile_dim2=0
    g1[5] = (int)(stride0);                           // tensor_dim0_stride lo
    g1[6] = 0;                                        // stride0 hi | stride1 lo
    g1[7] = 0;
    i32x4 z4 = {};
    i32x8 z8 = {};
    __builtin_amdgcn_tensor_load_to_lds(g0, g1, z4, z4, z8, 0);
}

// ---------------- conversion / transpose kernels ---------------------------
__global__ __launch_bounds__(256) void cvt_kernel(const float* __restrict__ src,
                                                  __bf16* __restrict__ dst, int n) {
    int i = blockIdx.x * 256 + threadIdx.x;
    if (i < n) dst[i] = (__bf16)src[i];
}
// WT[n][k] = W[k][n];  W is [rows=K][cols=N]
__global__ __launch_bounds__(256) void cvtT_kernel(const float* __restrict__ W,
                                                   __bf16* __restrict__ WT,
                                                   int rows, int cols) {
    int i = blockIdx.x * 256 + threadIdx.x;
    if (i >= rows * cols) return;
    int n = i / rows, k = i % rows;
    WT[(long)n * rows + k] = (__bf16)W[(long)k * cols + n];
}

// ---------------- Kernel 1: QKV projection  [4096x512]·[512x1536] ----------
__global__ __launch_bounds__(256) void qkv_kernel(const __bf16* __restrict__ xb,
                                                  const __bf16* __restrict__ WqkvT,
                                                  const float* __restrict__ bqkv,
                                                  __bf16* __restrict__ Qb,
                                                  __bf16* __restrict__ Kb,
                                                  __bf16* __restrict__ Vb) {
    int lane = threadIdx.x & 31, wave = threadIdx.x >> 5;
    int tile = blockIdx.x * 8 + wave;              // 256 m-tiles * 96 n-tiles
    int mt = tile / 96, nt = tile % 96;
    int m0 = mt * 16, n0 = nt * 16;
    v8f c = {};
#pragma unroll 4
    for (int k0 = 0; k0 < DIMX; k0 += 32) {
        v16bf a = load_a_bf16(xb, DIMX, m0, k0, lane);
        v16bf b = load_bT_bf16(WqkvT, DIMX, k0, n0, lane);
        c = WMMA_BF16(c, a, b);
    }
    int half = lane >> 4, n = lane & 15;
    int ncol  = n0 + n;
    float bias = bqkv[ncol];
    int which = ncol / DIMX;                       // 0=q 1=k 2=v
    int cc    = ncol % DIMX;
    int h = cc >> 6, d = cc & 63;
    __bf16* dst = (which == 0) ? Qb : ((which == 1) ? Kb : Vb);
#pragma unroll
    for (int r = 0; r < 8; ++r) {
        int m = m0 + r + 8 * half;                 // token index in [0,4096)
        int bi = m >> 9, s = m & 511;
        dst[(((long)bi * HEADS + h) * SEQ + s) * HEAD_DIM + d] = (__bf16)(c[r] + bias);
    }
}

// ---------------- Kernel 2: qp[b,h,i,t] = q . pos_emb_head(t) --------------
__global__ __launch_bounds__(256) void qp_kernel(const __bf16* __restrict__ Qb,
                                                 const float* __restrict__ pos_emb,
                                                 float* __restrict__ qp) {
    int lane = threadIdx.x & 31, wave = threadIdx.x >> 5;
    int tile = blockIdx.x * 8 + wave;              // 64 bh * (32 mt * 4 nt)
    int bh = tile >> 7;
    int t2 = tile & 127;
    int mt = t2 >> 2, nt = t2 & 3;
    int i0 = mt * 16, t0 = nt * 16;
    int h = bh & 7;
    const __bf16* Q = Qb + (long)bh * SEQ * HEAD_DIM;
    v8f c = {};
#pragma unroll
    for (int k0 = 0; k0 < HEAD_DIM; k0 += 32) {
        v16bf a = load_a_bf16(Q, HEAD_DIM, i0, k0, lane);
        v16bf b = load_bT_f32(pos_emb + h * HEAD_DIM, DIMX, k0, t0, lane);
        c = WMMA_BF16(c, a, b);
    }
    int half = lane >> 4, n = lane & 15;
#pragma unroll
    for (int r = 0; r < 8; ++r)
        qp[((long)bh * SEQ + i0 + r + 8 * half) * PTAB + t0 + n] = c[r];
}

// ---------------- Kernel 3: attention, 1 wave per (b,h, 16-row i-tile) -----
__global__ __launch_bounds__(32) void attn_kernel(const __bf16* __restrict__ Qb,
                                                  const __bf16* __restrict__ Kb,
                                                  const __bf16* __restrict__ Vb,
                                                  const float* __restrict__ qp,
                                                  __bf16* __restrict__ Ob) {
    __shared__ float  sc[16 * SEQ];          // 32 KB score tile
    __shared__ float  rowsum[16];
    __shared__ __bf16 ktile[16 * HEAD_DIM];  // 2 KB TDM-staged K tile
    __shared__ __bf16 vstage[32 * HEAD_DIM]; // 4 KB async-staged V tile

    int lane = threadIdx.x;
    int tile = blockIdx.x;                   // 64 bh * 32 i-tiles
    int bh = tile >> 5;
    int i0 = (tile & 31) * 16;
    int b  = bh >> 3, h = bh & 7;
    const __bf16* Q = Qb + (long)bh * SEQ * HEAD_DIM;
    const __bf16* K = Kb + (long)bh * SEQ * HEAD_DIM;
    const __bf16* V = Vb + (long)bh * SEQ * HEAD_DIM;
    const float* qprow = qp + (long)bh * SEQ * PTAB;
    const float scale = 0.125f;              // 1/sqrt(64)

    int half = lane >> 4, n = lane & 15;

    // Q fragments are invariant over the whole kernel
    v16bf aq0 = load_a_bf16(Q, HEAD_DIM, i0, 0,  lane);
    v16bf aq1 = load_a_bf16(Q, HEAD_DIM, i0, 32, lane);

    uint32_t ktile_lds = (uint32_t)(uintptr_t)ktile;

    // ---- scores: S = (Q K^T)*scale + qp[i, clip(j-i+32,0,63)] -------------
    for (int j0 = 0; j0 < SEQ; j0 += 16) {
        // prior LDS reads of ktile must retire before TDM overwrites (WAR)
        asm volatile("s_wait_dscnt 0x0" ::: "memory");
        // TDM: stage K[j0..j0+16)[0..64) bf16 (2 KB) into LDS
        tdm_load_2d_bf16(ktile_lds, K + (long)j0 * HEAD_DIM,
                         HEAD_DIM, SEQ, HEAD_DIM, HEAD_DIM, 16);
        __builtin_amdgcn_s_wait_tensorcnt(0);

        v8f c = {};
        // B[k=d][n=j] = ktile[j][d] : contiguous along k
        v16bf bk0 = load_bT_bf16(ktile, HEAD_DIM, 0,  0, lane);
        v16bf bk1 = load_bT_bf16(ktile, HEAD_DIM, 32, 0, lane);
        c = WMMA_BF16(c, aq0, bk0);
        c = WMMA_BF16(c, aq1, bk1);

        int j = j0 + n;
#pragma unroll
        for (int r = 0; r < 8; ++r) {
            int irel = r + 8 * half;
            int i = i0 + irel;
            int rp = j - i + 32;
            rp = rp < 0 ? 0 : (rp > 63 ? 63 : rp);
            sc[irel * SEQ + j] = c[r] * scale + qprow[(long)i * PTAB + rp];
        }
    }
    asm volatile("s_wait_dscnt 0x0" ::: "memory");
    __syncthreads();

    // ---- softmax over j (lane r owns row r) -------------------------------
    if (lane < 16) {
        float m = -3.0e38f;
        for (int j = 0; j < SEQ; ++j) m = fmaxf(m, sc[lane * SEQ + j]);
        float s = 0.0f;
        for (int j = 0; j < SEQ; ++j) {
            float e = __expf(sc[lane * SEQ + j] - m);
            sc[lane * SEQ + j] = e;
            s += e;
        }
        rowsum[lane] = s;
    }
    asm volatile("s_wait_dscnt 0x0" ::: "memory");
    __syncthreads();

    // ---- O = P V : V staged 32 rows at a time via ASYNC load-to-LDS -------
    v8f acc[4];
#pragma unroll
    for (int t = 0; t < 4; ++t) acc[t] = (v8f){};

    for (int j0 = 0; j0 < SEQ; j0 += 32) {
        asm volatile("s_wait_dscnt 0x0" ::: "memory");   // WAR on vstage
        const __bf16* vsrc = V + (long)j0 * HEAD_DIM;
        uint32_t lds_base = (uint32_t)(uintptr_t)vstage;
#pragma unroll
        for (int it = 0; it < 8; ++it) {
            uint32_t laddr = lds_base + (uint32_t)(lane * 16 + it * 512);
            uint64_t gaddr = (uint64_t)(uintptr_t)(vsrc + it * 256 + lane * 8);
            asm volatile("global_load_async_to_lds_b128 %0, %1, off"
                         :: "v"(laddr), "v"(gaddr) : "memory");
        }
        asm volatile("s_wait_asynccnt 0x0" ::: "memory");

        v16bf a = load_a_lds_f32(sc, SEQ, j0, lane);     // P tile (fp32->bf16)
#pragma unroll
        for (int t = 0; t < 4; ++t) {
            v16bf bm = load_b_lds_bf16(vstage, HEAD_DIM, t * 16, lane);
            acc[t] = WMMA_BF16(acc[t], a, bm);
        }
    }

    // write Ob[b][i][h*64+d] = acc / rowsum   (bf16 for the final projection)
#pragma unroll
    for (int t = 0; t < 4; ++t) {
#pragma unroll
        for (int r = 0; r < 8; ++r) {
            int irel = r + 8 * half;
            float inv = 1.0f / rowsum[irel];
            Ob[((long)b * SEQ + i0 + irel) * DIMX + h * HEAD_DIM + t * 16 + n] =
                (__bf16)(acc[t][r] * inv);
        }
    }
}

// ---------------- Kernel 4: output projection [4096x512]·[512x512] ---------
__global__ __launch_bounds__(256) void proj_kernel(const __bf16* __restrict__ Ob,
                                                   const __bf16* __restrict__ WoutT,
                                                   const float* __restrict__ bout,
                                                   float* __restrict__ out) {
    int lane = threadIdx.x & 31, wave = threadIdx.x >> 5;
    int tile = blockIdx.x * 8 + wave;              // 256 m-tiles * 32 n-tiles
    int mt = tile >> 5, nt = tile & 31;
    int m0 = mt * 16, n0 = nt * 16;
    v8f c = {};
#pragma unroll 4
    for (int k0 = 0; k0 < DIMX; k0 += 32) {
        v16bf a = load_a_bf16(Ob, DIMX, m0, k0, lane);
        v16bf b = load_bT_bf16(WoutT, DIMX, k0, n0, lane);
        c = WMMA_BF16(c, a, b);
    }
    int half = lane >> 4, n = lane & 15;
    float bias = bout[n0 + n];
#pragma unroll
    for (int r = 0; r < 8; ++r)
        out[(long)(m0 + r + 8 * half) * DIMX + n0 + n] = c[r] + bias;
}

// ---------------- launcher -------------------------------------------------
extern "C" void kernel_launch(void* const* d_in, const int* in_sizes, int n_in,
                              void* d_out, int out_size, void* d_ws, size_t ws_size,
                              hipStream_t stream) {
    const float* x    = (const float*)d_in[0];
    const float* Wqkv = (const float*)d_in[1];
    const float* bqkv = (const float*)d_in[2];
    const float* Wout = (const float*)d_in[3];
    const float* bout = (const float*)d_in[4];
    const float* pos  = (const float*)d_in[5];

    char* ws = (char*)d_ws;
    __bf16* Qb    = (__bf16*)(ws);                        // 4 MiB
    __bf16* Kb    = (__bf16*)(ws + ((size_t)4  << 20));   // 4 MiB
    __bf16* Vb    = (__bf16*)(ws + ((size_t)8  << 20));   // 4 MiB
    float*  qp    = (float*) (ws + ((size_t)12 << 20));   // 8 MiB
    __bf16* Ob    = (__bf16*)(ws + ((size_t)20 << 20));   // 4 MiB
    __bf16* xb    = (__bf16*)(ws + ((size_t)24 << 20));   // 4 MiB
    __bf16* WqkvT = (__bf16*)(ws + ((size_t)28 << 20));   // 1.5 MiB
    __bf16* WoutT = (__bf16*)(ws + ((size_t)30 << 20));   // 0.5 MiB
    float*  out   = (float*)d_out;

    cvt_kernel <<<8192, 256, 0, stream>>>(x, xb, BATCH * SEQ * DIMX);
    cvtT_kernel<<<3072, 256, 0, stream>>>(Wqkv, WqkvT, DIMX, 3 * DIMX);
    cvtT_kernel<<<1024, 256, 0, stream>>>(Wout, WoutT, DIMX, DIMX);

    qkv_kernel <<<3072, 256, 0, stream>>>(xb, WqkvT, bqkv, Qb, Kb, Vb);
    qp_kernel  <<<1024, 256, 0, stream>>>(Qb, pos, qp);
    attn_kernel<<<2048,  32, 0, stream>>>(Qb, Kb, Vb, qp, Ob);
    proj_kernel<<<1024, 256, 0, stream>>>(Ob, WoutT, bout, out);
}
